// GAT1_31619549233397
// MI455X (gfx1250) — compile-verified, matrix operand order checked by reading, not run on hardware
//
#include <hip/hip_runtime.h>
#include <hip/hip_bf16.h>
#include <math.h>

// ---------------------------------------------------------------------------
// Problem constants (match reference)
// ---------------------------------------------------------------------------
constexpr int N_  = 100000;
constexpr int E_  = 400000;
constexpr int B_  = 2000;
constexpr int G_  = 50;
constexpr int H_  = 4;
constexpr int D_  = 64;
constexpr int HD_ = 256;          // H*D
constexpr int FIN_   = 74;
constexpr int KP1_   = 96;        // FIN padded to mult of 32
constexpr int FP_    = 5000;
constexpr int HID_   = 512;
constexpr int OUT_   = 2;
constexpr int MLPIN_ = D_ + FP_;  // 5064
constexpr int KPM_   = 5088;      // MLPIN padded to mult of 32
constexpr float ELU_A_   = 0.2f;
constexpr float LRELU_A_ = 0.2f;

typedef __bf16 bf16_t;
typedef bf16_t v16bf __attribute__((ext_vector_type(16)));
typedef bf16_t v8bf  __attribute__((ext_vector_type(8)));
typedef float  v8f   __attribute__((ext_vector_type(8)));

static inline int ceildiv(long long a, long long b) { return (int)((a + b - 1) / b); }

// float -> bf16 (round-to-nearest-even), stored as u16 to avoid relying on
// scalar __bf16 conversions in every kernel
__device__ __forceinline__ unsigned short f2bf(float f) {
  unsigned u = __float_as_uint(f);
  unsigned r = u + 0x7FFFu + ((u >> 16) & 1u);
  return (unsigned short)(r >> 16);
}

__device__ __forceinline__ v16bf cat8(v8bf lo, v8bf hi) {
  return __builtin_shufflevector(lo, hi, 0,1,2,3,4,5,6,7,8,9,10,11,12,13,14,15);
}

// ---------------------------------------------------------------------------
// Generic bf16 WMMA GEMM:  C[M,Ncols](f32) = A[M,Kpad](bf16) * Bt[Ncols,Kpad]^T
// Each wave computes a 16x64 tile (4 WMMAs per 32-wide K step, A reused).
// A fragment layout (16-bit A, 16x32): lanes 0-15 -> row M=lane, K = kb+{0..7,16..23}
//                                      lanes16-31 -> row M=lane-16, K = kb+{8..15,24..31}
// B fragment symmetric with N in place of M.  C/D: VGPR r, lane<16 -> M=r, N=lane;
// lane>=16 -> M=r+8, N=lane-16.
// MODE 0: plain store.  MODE 1: relu(acc+bias)*gamma*bnInv + beta (eval BatchNorm).
// ---------------------------------------------------------------------------
template<int MODE>
__global__ __launch_bounds__(256)
void wmma_gemm_kernel(const unsigned short* __restrict__ Au,
                      const unsigned short* __restrict__ Btu,
                      float* __restrict__ C,
                      int M, int Ncols, int Kpad,
                      const float* __restrict__ bias,
                      const float* __restrict__ gamma,
                      const float* __restrict__ beta,
                      float bnInv) {
  const bf16_t* A  = reinterpret_cast<const bf16_t*>(Au);
  const bf16_t* Bt = reinterpret_cast<const bf16_t*>(Btu);

  const int lane   = threadIdx.x & 31;
  const int waveId = threadIdx.x >> 5;
  const int colGroups = Ncols >> 6;               // tiles of 64 columns
  const int rowTiles  = M >> 4;
  int tile = blockIdx.x * (blockDim.x >> 5) + waveId;
  if (tile >= rowTiles * colGroups) return;       // wave-uniform exit
  const int rowT = tile / colGroups;
  const int colG = tile % colGroups;
  const int row0 = rowT << 4;
  const int col0 = colG << 6;

  const int laneLo = lane & 15;
  const int kOff   = (lane >> 4) << 3;            // 0 for lanes 0-15, 8 for 16-31

  v8f acc[4] = {};

  const bf16_t* aRow = A + (size_t)(row0 + laneLo) * Kpad;
  for (int kb = 0; kb < Kpad; kb += 32) {
    const bf16_t* ap = aRow + kb + kOff;
    v16bf a = cat8(*(const v8bf*)ap, *(const v8bf*)(ap + 16));
#pragma unroll
    for (int j = 0; j < 4; ++j) {
      const bf16_t* bp = Bt + (size_t)(col0 + (j << 4) + laneLo) * Kpad + kb + kOff;
      v16bf b = cat8(*(const v8bf*)bp, *(const v8bf*)(bp + 16));
      acc[j] = __builtin_amdgcn_wmma_f32_16x16x32_bf16(
          /*neg_a=*/false, a, /*neg_b=*/false, b,
          /*c_mod=*/(short)0, acc[j], /*reuse_a=*/false, /*reuse_b=*/false);
    }
  }

  const int rowAdd = (lane >> 4) << 3;            // +8 rows for lanes 16-31
#pragma unroll
  for (int j = 0; j < 4; ++j) {
    const int cc = col0 + (j << 4) + laneLo;
    float bia = 0.f, sc = 1.f, sh = 0.f;
    if (MODE == 1) { bia = bias[cc]; sc = gamma[cc] * bnInv; sh = beta[cc]; }
#pragma unroll
    for (int r = 0; r < 8; ++r) {
      const int rr = row0 + r + rowAdd;
      float v = acc[j][r];
      if (MODE == 1) v = fmaxf(v + bia, 0.f) * sc + sh;
      C[(size_t)rr * Ncols + cc] = v;
    }
  }
}

// ---------------------------------------------------------------------------
// Conversion / layout kernels
// ---------------------------------------------------------------------------
__global__ void cvt_pad_bf16(const float* __restrict__ in, unsigned short* __restrict__ out,
                             int K, int Kpad, long long total) {
  long long i = (long long)blockIdx.x * blockDim.x + threadIdx.x;
  if (i >= total) return;
  int k = (int)(i % Kpad);
  long long m = i / Kpad;
  out[i] = (k < K) ? f2bf(in[m * K + k]) : (unsigned short)0;
}

// W[K,Ncols] (f32, row-major) -> Wt[Ncols,Kpad] (bf16), zero-padded
__global__ void cvt_wT_bf16(const float* __restrict__ W, unsigned short* __restrict__ Wt,
                            int K, int Ncols, int Kpad, long long total) {
  long long i = (long long)blockIdx.x * blockDim.x + threadIdx.x;
  if (i >= total) return;
  int k = (int)(i % Kpad);
  int n = (int)(i / Kpad);
  Wt[i] = (k < K) ? f2bf(W[(size_t)k * Ncols + n]) : (unsigned short)0;
}

// concat [gh | grover_fp | pad] -> bf16 [B, KPM]
__global__ void concat_bf16(const float* __restrict__ gh, const float* __restrict__ fp,
                            unsigned short* __restrict__ out, long long total) {
  long long i = (long long)blockIdx.x * blockDim.x + threadIdx.x;
  if (i >= total) return;
  int k = (int)(i % KPM_);
  int b = (int)(i / KPM_);
  float v = 0.f;
  if (k < D_)            v = gh[b * D_ + k];
  else if (k < MLPIN_)   v = fp[(size_t)b * FP_ + (k - D_)];
  out[i] = f2bf(v);
}

__global__ void fill_f32(float* __restrict__ p, float v, long long n) {
  long long i = (long long)blockIdx.x * blockDim.x + threadIdx.x;
  if (i < n) p[i] = v;
}

// ---------------------------------------------------------------------------
// GAT attention kernels
// ---------------------------------------------------------------------------
// el[n,h] = sum_d ft[n,h*64+d]*al[h,d]; er likewise. One wave per (n,h).
__global__ void attn_coef(const float* __restrict__ ft, const float* __restrict__ al,
                          const float* __restrict__ ar, float* __restrict__ el,
                          float* __restrict__ er) {
  int wid  = (int)(((long long)blockIdx.x * blockDim.x + threadIdx.x) >> 5);
  int lane = threadIdx.x & 31;
  if (wid >= N_ * H_) return;
  int n = wid >> 2, h = wid & 3;
  const float* f = ft + (size_t)n * HD_ + h * D_;
  const float* a = al + h * D_;
  const float* b = ar + h * D_;
  float sl = f[lane] * a[lane] + f[lane + 32] * a[lane + 32];
  float sr = f[lane] * b[lane] + f[lane + 32] * b[lane + 32];
#pragma unroll
  for (int off = 16; off > 0; off >>= 1) {
    sl += __shfl_xor(sl, off, 32);
    sr += __shfl_xor(sr, off, 32);
  }
  if (lane == 0) { el[wid] = sl; er[wid] = sr; }
}

__device__ __forceinline__ void atomicMaxFloat(float* addr, float val) {
  if (val >= 0.f) atomicMax((int*)addr, __float_as_int(val));
  else            atomicMin((unsigned int*)addr, __float_as_uint(val));
}

// pass 1: e = leaky_relu(el[src]+er[dst]); segment max over dst
__global__ void edge_logits(const int* __restrict__ src, const int* __restrict__ dst,
                            const float* __restrict__ el, const float* __restrict__ er,
                            float* __restrict__ ebuf, float* __restrict__ m) {
  long long i = (long long)blockIdx.x * blockDim.x + threadIdx.x;
  if (i >= (long long)E_ * H_) return;
  int e = (int)(i >> 2), h = (int)(i & 3);
  int s = src[e], d = dst[e];
  float x = el[s * H_ + h] + er[d * H_ + h];
  x = (x > 0.f) ? x : LRELU_A_ * x;
  ebuf[i] = x;
  atomicMaxFloat(&m[d * H_ + h], x);
}

// pass 2: ex = exp(e - m[dst]); segment sum over dst
__global__ void edge_exp(const int* __restrict__ dst, const float* __restrict__ m,
                         float* __restrict__ ebuf, float* __restrict__ den) {
  long long i = (long long)blockIdx.x * blockDim.x + threadIdx.x;
  if (i >= (long long)E_ * H_) return;
  int e = (int)(i >> 2), h = (int)(i & 3);
  int d = dst[e];
  float ex = __expf(ebuf[i] - m[d * H_ + h]);
  ebuf[i] = ex;
  atomicAdd(&den[d * H_ + h], ex);
}

// pass 3: rst[dst,h,:] += ft[src,h,:] * (ex/den[dst,h]); one wave per (e,h)
__global__ void edge_aggregate(const int* __restrict__ src, const int* __restrict__ dst,
                               const float* __restrict__ ebuf, const float* __restrict__ den,
                               const float* __restrict__ ft, float* __restrict__ rst) {
  long long wid = ((long long)blockIdx.x * blockDim.x + threadIdx.x) >> 5;
  int lane = threadIdx.x & 31;
  if (wid >= (long long)E_ * H_) return;
  int e = (int)(wid >> 2), h = (int)(wid & 3);
  int s = src[e], d = dst[e];
  float a = ebuf[wid] / den[d * H_ + h];
  const float* f = ft + (size_t)s * HD_ + h * D_;
  float*       o = rst + (size_t)d * HD_ + h * D_;
  atomicAdd(&o[lane],      f[lane]      * a);
  atomicAdd(&o[lane + 32], f[lane + 32] * a);
}

// post for layers 1,2: out = elu(rst + b + res)  (flatten agg)
__global__ void post_elu(const float* __restrict__ rst, const float* __restrict__ b,
                         const float* __restrict__ res, float* __restrict__ out) {
  long long i = (long long)blockIdx.x * blockDim.x + threadIdx.x;
  if (i >= (long long)N_ * HD_) return;
  int c = (int)(i & (HD_ - 1));
  float x = rst[i] + b[c] + res[i];
  out[i] = (x > 0.f) ? x : ELU_A_ * (__expf(x) - 1.f);
}

// post for layer 3: out[n,d] = mean_h(rst + b + res)
__global__ void post_mean(const float* __restrict__ rst, const float* __restrict__ b,
                          const float* __restrict__ res, float* __restrict__ out) {
  long long i = (long long)blockIdx.x * blockDim.x + threadIdx.x;
  if (i >= (long long)N_ * D_) return;
  int d = (int)(i & (D_ - 1));
  long long n = i >> 6;
  float s = 0.f;
#pragma unroll
  for (int h = 0; h < H_; ++h) {
    size_t idx = (size_t)n * HD_ + h * D_ + d;
    s += rst[idx] + b[h * D_ + d] + res[idx];
  }
  out[i] = s * 0.25f;
}

// per-graph mean pool (nodes are contiguous: graph g owns [g*G, g*G+G))
__global__ void pool_mean(const float* __restrict__ hf, float* __restrict__ gh) {
  long long i = (long long)blockIdx.x * blockDim.x + threadIdx.x;
  if (i >= (long long)B_ * D_) return;
  int d = (int)(i & (D_ - 1));
  int g = (int)(i >> 6);
  float s = 0.f;
  for (int j = 0; j < G_; ++j) s += hf[(size_t)(g * G_ + j) * D_ + d];
  gh[i] = s * (1.f / (float)G_);
}

// tiny output head: out[b,o] = x2[b,:] . Wo[:,o] + bo[o]
__global__ void final_out(const float* __restrict__ x2, const float* __restrict__ Wo,
                          const float* __restrict__ bo, float* __restrict__ out) {
  long long i = (long long)blockIdx.x * blockDim.x + threadIdx.x;
  if (i >= (long long)B_ * OUT_) return;
  int o = (int)(i % OUT_);
  int b = (int)(i / OUT_);
  float s = bo[o];
  const float* x = x2 + (size_t)b * HID_;
  for (int k = 0; k < HID_; ++k) s += x[k] * Wo[k * OUT_ + o];
  out[i] = s;
}

// ---------------------------------------------------------------------------
// Orchestration
// ---------------------------------------------------------------------------
extern "C" void kernel_launch(void* const* d_in, const int* in_sizes, int n_in,
                              void* d_out, int out_size, void* d_ws, size_t ws_size,
                              hipStream_t stream) {
  (void)in_sizes; (void)n_in; (void)out_size; (void)ws_size;

  // ---- inputs (dict/insertion-order flattening; Wres=None skipped) ----
  const float* node_h    = (const float*)d_in[0];   // [N,74]
  const int*   src       = (const int*)  d_in[1];   // [E]
  const int*   dst       = (const int*)  d_in[2];   // [E]
  /* graph_id d_in[3] unused: nodes contiguous per graph */
  const float* grover_fp = (const float*)d_in[4];   // [B,5000]
  const float* W0    = (const float*)d_in[5];       // [74,256]
  const float* al0   = (const float*)d_in[6];
  const float* ar0   = (const float*)d_in[7];
  const float* b0    = (const float*)d_in[8];
  const float* Wres0 = (const float*)d_in[9];       // [74,256]
  const float* W1g   = (const float*)d_in[10];      // [256,256]
  const float* al1   = (const float*)d_in[11];
  const float* ar1   = (const float*)d_in[12];
  const float* b1g   = (const float*)d_in[13];
  const float* W2g   = (const float*)d_in[14];
  const float* al2   = (const float*)d_in[15];
  const float* ar2   = (const float*)d_in[16];
  const float* b2g   = (const float*)d_in[17];
  const float* mW1   = (const float*)d_in[18];      // [5064,512]
  const float* mb1   = (const float*)d_in[19];
  const float* mg1   = (const float*)d_in[20];
  const float* mbe1  = (const float*)d_in[21];
  const float* mW2   = (const float*)d_in[22];      // [512,512]
  const float* mb2   = (const float*)d_in[23];
  const float* mg2   = (const float*)d_in[24];
  const float* mbe2  = (const float*)d_in[25];
  const float* mWo   = (const float*)d_in[26];      // [512,2]
  const float* mbo   = (const float*)d_in[27];
  float* out = (float*)d_out;                       // [B,2]

  // ---- workspace carve (256B aligned) ----
  char* ws = (char*)d_ws;
  size_t off = 0;
  auto carve = [&](size_t bytes) { void* p = ws + off; off += (bytes + 255) & ~(size_t)255; return p; };
  float* FT   = (float*)carve((size_t)N_ * HD_ * 4);   // per-layer ft; HF aliases its head
  float* RST  = (float*)carve((size_t)N_ * HD_ * 4);   // per-layer agg; ABF/XIN alias its head
  float* H1   = (float*)carve((size_t)N_ * HD_ * 4);
  float* H2   = (float*)carve((size_t)N_ * HD_ * 4);   // layer-1 residual, then layer-2 out
  unsigned short* WT0 = (unsigned short*)carve((size_t)HID_ * KPM_ * 2);
  unsigned short* WT1 = (unsigned short*)carve((size_t)HD_ * HD_ * 2);
  float* EL   = (float*)carve((size_t)N_ * H_ * 4);
  float* ER   = (float*)carve((size_t)N_ * H_ * 4);
  float* MB   = (float*)carve((size_t)N_ * H_ * 4);
  float* DEN  = (float*)carve((size_t)N_ * H_ * 4);
  float* EBUF = (float*)carve((size_t)E_ * H_ * 4);
  float* GH   = (float*)carve((size_t)B_ * D_ * 4);
  float* X1   = (float*)carve((size_t)B_ * HID_ * 4);
  unsigned short* X1BF = (unsigned short*)carve((size_t)B_ * HID_ * 2);
  float* X2   = (float*)carve((size_t)B_ * HID_ * 4);
  unsigned short* ABF = (unsigned short*)RST;          // bf16 activations (GEMM-time only)
  float* HF = FT;                                      // [N,64] final node feats

  const float bnInv = 1.0f / sqrtf(1.0f + 1e-5f);
  const int T = 256;
  auto g1d = [&](long long n) { return dim3((unsigned)ceildiv(n, T)); };

  long long nhTot = (long long)N_ * H_;
  long long ehTot = (long long)E_ * H_;
  long long nhdTot = (long long)N_ * HD_;
  dim3 gemmGridL(((N_ / 16) * (HD_ / 64)) / 8);        // 3125 blocks, 8 waves each
  dim3 gemmGridM(((B_ / 16) * (HID_ / 64)) / 8);       // 125 blocks

  auto run_edges = [&](const float* al, const float* ar, const float* bvec,
                       const float* res, float* outh, bool mean) {
    attn_coef<<<g1d(nhTot * 32), T, 0, stream>>>(FT, al, ar, EL, ER);
    fill_f32<<<g1d(nhTot), T, 0, stream>>>(MB, -3.0e38f, nhTot);
    fill_f32<<<g1d(nhTot), T, 0, stream>>>(DEN, 0.f, nhTot);
    fill_f32<<<g1d(nhdTot), T, 0, stream>>>(RST, 0.f, nhdTot);
    edge_logits<<<g1d(ehTot), T, 0, stream>>>(src, dst, EL, ER, EBUF, MB);
    edge_exp<<<g1d(ehTot), T, 0, stream>>>(dst, MB, EBUF, DEN);
    edge_aggregate<<<g1d(ehTot * 32), T, 0, stream>>>(src, dst, EBUF, DEN, FT, RST);
    if (!mean) post_elu<<<g1d(nhdTot), T, 0, stream>>>(RST, bvec, res, outh);
    else       post_mean<<<g1d((long long)N_ * D_), T, 0, stream>>>(RST, bvec, res, outh);
  };

  // ================= GAT layer 1 (K=74 -> pad 96) =================
  {
    long long aTot = (long long)N_ * KP1_, wTot = (long long)HD_ * KP1_;
    cvt_pad_bf16<<<g1d(aTot), T, 0, stream>>>(node_h, ABF, FIN_, KP1_, aTot);
    cvt_wT_bf16<<<g1d(wTot), T, 0, stream>>>(W0, WT1, FIN_, HD_, KP1_, wTot);
    wmma_gemm_kernel<0><<<gemmGridL, T, 0, stream>>>(ABF, WT1, FT, N_, HD_, KP1_,
                                                     nullptr, nullptr, nullptr, 0.f);
    cvt_wT_bf16<<<g1d(wTot), T, 0, stream>>>(Wres0, WT1, FIN_, HD_, KP1_, wTot);
    wmma_gemm_kernel<0><<<gemmGridL, T, 0, stream>>>(ABF, WT1, H2, N_, HD_, KP1_,
                                                     nullptr, nullptr, nullptr, 0.f);
    run_edges(al0, ar0, b0, /*res=*/H2, /*out=*/H1, /*mean=*/false);
  }
  // ================= GAT layer 2 (K=256) =================
  {
    long long wTot = (long long)HD_ * HD_;
    cvt_pad_bf16<<<g1d(nhdTot), T, 0, stream>>>(H1, ABF, HD_, HD_, nhdTot);
    cvt_wT_bf16<<<g1d(wTot), T, 0, stream>>>(W1g, WT1, HD_, HD_, HD_, wTot);
    wmma_gemm_kernel<0><<<gemmGridL, T, 0, stream>>>(ABF, WT1, FT, N_, HD_, HD_,
                                                     nullptr, nullptr, nullptr, 0.f);
    run_edges(al1, ar1, b1g, /*res=*/H1, /*out=*/H2, /*mean=*/false);
  }
  // ================= GAT layer 3 (K=256, head-mean) =================
  {
    long long wTot = (long long)HD_ * HD_;
    cvt_pad_bf16<<<g1d(nhdTot), T, 0, stream>>>(H2, ABF, HD_, HD_, nhdTot);
    cvt_wT_bf16<<<g1d(wTot), T, 0, stream>>>(W2g, WT1, HD_, HD_, HD_, wTot);
    wmma_gemm_kernel<0><<<gemmGridL, T, 0, stream>>>(ABF, WT1, FT, N_, HD_, HD_,
                                                     nullptr, nullptr, nullptr, 0.f);
    run_edges(al2, ar2, b2g, /*res=*/H2, /*out=*/HF, /*mean=*/true);
  }
  // ================= pooling + MLP =================
  pool_mean<<<g1d((long long)B_ * D_), T, 0, stream>>>(HF, GH);
  {
    unsigned short* XIN = ABF;                 // RST free now
    long long xTot = (long long)B_ * KPM_, w1Tot = (long long)HID_ * KPM_;
    concat_bf16<<<g1d(xTot), T, 0, stream>>>(GH, grover_fp, XIN, xTot);
    cvt_wT_bf16<<<g1d(w1Tot), T, 0, stream>>>(mW1, WT0, MLPIN_, HID_, KPM_, w1Tot);
    wmma_gemm_kernel<1><<<gemmGridM, T, 0, stream>>>(XIN, WT0, X1, B_, HID_, KPM_,
                                                     mb1, mg1, mbe1, bnInv);
    long long x1Tot = (long long)B_ * HID_, w2Tot = (long long)HID_ * HID_;
    cvt_pad_bf16<<<g1d(x1Tot), T, 0, stream>>>(X1, X1BF, HID_, HID_, x1Tot);
    cvt_wT_bf16<<<g1d(w2Tot), T, 0, stream>>>(mW2, WT0, HID_, HID_, HID_, w2Tot);
    wmma_gemm_kernel<1><<<gemmGridM, T, 0, stream>>>(X1BF, WT0, X2, B_, HID_, HID_,
                                                     mb2, mg2, mbe2, bnInv);
    final_out<<<g1d((long long)B_ * OUT_), T, 0, stream>>>(X2, mWo, mbo, out);
  }
}
